// FairseqHuBERT_72919954751847
// MI455X (gfx1250) — matrix operand order repaired and checked by reading, not compiled, not run on hardware
//
#include <hip/hip_runtime.h>
#include <hip/hip_bf16.h>

typedef __attribute__((ext_vector_type(16))) __bf16 v16bf;
typedef __attribute__((ext_vector_type(8)))  float  v8f;
typedef __attribute__((ext_vector_type(4)))  float  v4f;

#define T_FRAMES 100000
#define D_DIM    768
#define K_CL     500
#define KP       512          // padded columns (32 tiles of 16)
#define BLK_ROWS 64           // 4 row-tiles of 16 per workgroup
#define DCHUNK   32           // WMMA K per step
#define NCHUNK   (D_DIM / DCHUNK)          // 24
#define CHUNK_HW (32 * 512)                // halfwords per bfC chunk (32 tiles * 512)

__device__ __forceinline__ unsigned f32_to_bf16_rne(float f) {
    unsigned u = __float_as_uint(f);
    return (u + 0x7FFFu + ((u >> 16) & 1u)) >> 16;   // round-to-nearest-even
}

// Monotonic order-preserving float->uint map packed with column index.
// min() over keys == (min score, then lowest index) — matches jnp.argmin ties.
__device__ __forceinline__ unsigned long long packKey(float v, unsigned idx) {
    unsigned u = __float_as_uint(v);
    u = (u & 0x80000000u) ? ~u : (u | 0x80000000u);
    return ((unsigned long long)u << 32) | (unsigned long long)idx;
}

// ---- pre-pass 1: Cnorm[k] = sum_d C[d][k]^2  (padded cols -> +inf) ----------
__global__ void cnorm_kernel(const float* __restrict__ C, float* __restrict__ cnorm) {
    int k = blockIdx.x * blockDim.x + threadIdx.x;
    if (k >= KP) return;
    if (k >= K_CL) { cnorm[k] = __builtin_inff(); return; }
    float s = 0.f;
    for (int d = 0; d < D_DIM; ++d) {
        float v = C[(size_t)d * K_CL + k];
        s = fmaf(v, v, s);
    }
    cnorm[k] = s;
}

// ---- pre-pass 2: convert C (f32 [D,K]) -> bf16 in WMMA B-fragment layout ----
// bfC[chunk][ct][lane][reg*2+hw]; contiguous k order chosen here (not AMD table order): lanes 0-15 K=0..15 (N=lane),
// lanes 16-31 K=16..31; halfword pairs per dword.
__global__ void convC_kernel(const float* __restrict__ C, unsigned short* __restrict__ bfC) {
    int e = blockIdx.x * blockDim.x + threadIdx.x;     // NCHUNK*32*512 = 393216
    if (e >= NCHUNK * 32 * KP) return;
    int kc = e & 511;            // column 0..511
    int kr = (e >> 9) & 31;      // k within chunk
    int c  = e >> 14;            // chunk
    float v = (kc < K_CL) ? C[(size_t)(c * DCHUNK + kr) * K_CL + kc] : 0.f;
    int n = kc & 15, ct = kc >> 4;
    int hi = kr >> 4, reg = (kr & 15) >> 1, hw = kr & 1;
    int l = hi * 16 + n;
    bfC[(size_t)c * CHUNK_HW + ct * 512 + l * 16 + reg * 2 + hw] =
        (unsigned short)f32_to_bf16_rne(v);
}

// ---- main: bf16 WMMA GEMM tile + fused argmin, pipelined A staging ----------
__global__ void __launch_bounds__(256)
hubert_kmeans_kernel(const float* __restrict__ feat,
                     const unsigned short* __restrict__ bfC,
                     const float* __restrict__ cnorm,
                     int* __restrict__ out)
{
    __shared__ __align__(32) unsigned short ldsA[2 * 4 * 512];  // double buffer, 8 KB
    __shared__ float               ldsCn[KP];                   // 2 KB
    __shared__ unsigned long long  ldsBest[BLK_ROWS * 8];       // 4 KB

    const int tid  = threadIdx.x;
    const int wave = tid >> 5;        // 0..7, owns col tiles [wave*4, wave*4+4)
    const int lane = tid & 31;
    const int row0 = blockIdx.x * BLK_ROWS;

    for (int k = tid; k < KP; k += 256) ldsCn[k] = cnorm[k];

    // Per-thread A staging geometry (constant across chunks).
    // A 16x32 layout (matches bfC): lane l<16 row M=l, K 0..15;
    // lane>=16 same rows, K 16..31. 4 consecutive K (kq%4==0) map to
    // contiguous halfwords: offset = k&15.
    int srow[2], shwi[2], skq[2];
    #pragma unroll
    for (int i = 0; i < 2; ++i) {
        int u  = tid + i * 256;            // 512 float4-units per chunk
        int rl = u >> 3, kq = (u & 7) * 4;
        int row = row0 + rl; if (row >= T_FRAMES) row = T_FRAMES - 1;
        srow[i] = row;
        skq[i]  = kq;
        int m = rl & 15, rt = rl >> 4;
        int hi = kq >> 4, grp = (kq & 15) >> 3;
        shwi[i] = rt * 512 + ((hi * 16 + m) * 16) + grp * 8 + (kq & 7);
    }

    v8f acc[4][4] = {};   // [row-tile][col-tile] 16x16 f32 accumulators

    // ---- prologue: stage chunk 0 into buffer 0 ----
    v4f sfv[2];
    #pragma unroll
    for (int i = 0; i < 2; ++i)
        sfv[i] = __builtin_nontemporal_load(reinterpret_cast<const v4f*>(
                     feat + (size_t)srow[i] * D_DIM + 0 + skq[i]));
    #pragma unroll
    for (int i = 0; i < 2; ++i) {
        uint2 p;
        p.x = f32_to_bf16_rne(sfv[i].x) | (f32_to_bf16_rne(sfv[i].y) << 16);
        p.y = f32_to_bf16_rne(sfv[i].z) | (f32_to_bf16_rne(sfv[i].w) << 16);
        *reinterpret_cast<uint2*>(&ldsA[shwi[i]]) = p;     // ds_store_b64
    }
    __syncthreads();

    for (int c = 0; c < NCHUNK; ++c) {
        const int cur = c & 1;

        // issue feat loads for chunk c+1 early; latency hidden under compute
        if (c + 1 < NCHUNK) {
            int d0n = (c + 1) * DCHUNK;
            #pragma unroll
            for (int i = 0; i < 2; ++i)
                sfv[i] = __builtin_nontemporal_load(reinterpret_cast<const v4f*>(
                             feat + (size_t)srow[i] * D_DIM + d0n + skq[i]));
        }

        // ---- compute chunk c: A frags from LDS, B frags from global (L2) ----
        const unsigned short* bbase = bfC + (size_t)c * CHUNK_HW;
        v16bf afrag[4];
        #pragma unroll
        for (int rt = 0; rt < 4; ++rt)
            afrag[rt] = *reinterpret_cast<const v16bf*>(
                &ldsA[cur * 2048 + rt * 512 + lane * 16]);

        #pragma unroll
        for (int cl = 0; cl < 4; ++cl) {
            int ct = wave * 4 + cl;
            v16bf bfrag = *reinterpret_cast<const v16bf*>(bbase + ct * 512 + lane * 16);
            #pragma unroll
            for (int rt = 0; rt < 4; ++rt) {
                acc[rt][cl] = __builtin_amdgcn_wmma_f32_16x16x32_bf16(
                    false, afrag[rt], false, bfrag,
                    (short)0, acc[rt][cl], false, false);
            }
        }

        // ---- stage chunk c+1 into the other buffer; single barrier per iter
        if (c + 1 < NCHUNK) {
            #pragma unroll
            for (int i = 0; i < 2; ++i) {
                uint2 p;
                p.x = f32_to_bf16_rne(sfv[i].x) | (f32_to_bf16_rne(sfv[i].y) << 16);
                p.y = f32_to_bf16_rne(sfv[i].z) | (f32_to_bf16_rne(sfv[i].w) << 16);
                *reinterpret_cast<uint2*>(&ldsA[(1 - cur) * 2048 + shwi[i]]) = p;
            }
            __syncthreads();
        }
    }

    // ---- epilogue: score = Cnorm[col] - 2*dot ; fused argmin ---------------
    // C/D layout: VGPR j, lanes 0-15 -> (M=j, N=lane), lanes 16-31 -> (M=j+8)
    const int nl   = lane & 15;
    const int half = lane >> 4;
    #pragma unroll
    for (int rt = 0; rt < 4; ++rt) {
        #pragma unroll
        for (int j = 0; j < 8; ++j) {
            unsigned long long key = ~0ull;
            #pragma unroll
            for (int cl = 0; cl < 4; ++cl) {
                int col = wave * 64 + cl * 16 + nl;
                float score = ldsCn[col] - 2.0f * acc[rt][cl][j];
                unsigned long long k2 = packKey(score, (unsigned)col);
                key = (k2 < key) ? k2 : key;
            }
            #pragma unroll
            for (int m = 1; m < 16; m <<= 1) {     // reduce 16-lane halves
                unsigned long long o = __shfl_xor(key, m, 32);
                key = (o < key) ? o : key;
            }
            if (nl == 0) {
                int rlocal = rt * 16 + j + 8 * half;
                ldsBest[rlocal * 8 + wave] = key;
            }
        }
    }
    __syncthreads();

    if (tid < BLK_ROWS) {
        unsigned long long key = ldsBest[tid * 8];
        #pragma unroll
        for (int w = 1; w < 8; ++w) {
            unsigned long long o = ldsBest[tid * 8 + w];
            key = (o < key) ? o : key;
        }
        int row = row0 + tid;
        if (row < T_FRAMES)
            __builtin_nontemporal_store((int)(key & 0xFFFFFFFFull), out + row);
    }
}

extern "C" void kernel_launch(void* const* d_in, const int* in_sizes, int n_in,
                              void* d_out, int out_size, void* d_ws, size_t ws_size,
                              hipStream_t stream) {
    const float* feat = (const float*)d_in[0];   // [T, D] f32
    const float* C    = (const float*)d_in[1];   // [D, K] f32
    int*   out        = (int*)d_out;             // [1, T] int32 cluster ids

    // workspace layout: [0,2KB) cnorm f32[512]; [2KB, 2KB+768KB) bfC fragments
    float* cnorm        = (float*)d_ws;
    unsigned short* bfC = (unsigned short*)((char*)d_ws + 2048);

    cnorm_kernel<<<2, 256, 0, stream>>>(C, cnorm);

    int convElems = NCHUNK * 32 * KP;            // 393216
    convC_kernel<<<(convElems + 255) / 256, 256, 0, stream>>>(C, bfC);

    int nblk = (T_FRAMES + BLK_ROWS - 1) / BLK_ROWS;
    hubert_kmeans_kernel<<<nblk, 256, 0, stream>>>(feat, bfC, cnorm, out);
}
